// ScaledDotProductAttention_8237747273924
// MI455X (gfx1250) — matrix operand order repaired and checked by reading, not compiled, not run on hardware
//
#include <hip/hip_runtime.h>
#include <math.h>

// ---- CDNA5 WMMA vector types (gfx1250, wave32) ----
typedef __attribute__((ext_vector_type(16))) _Float16 v16h;
typedef __attribute__((ext_vector_type(16))) float    v16f;
typedef __attribute__((ext_vector_type(8)))  float    v8f;

#define B_  8
#define S_  2048
#define D_  128
#define QT  16   // query rows per workgroup

// LDS layout (dynamic): fp32 scores [16][2048] | fp16 weights [16][2048] | red [16][16]
#define SMEM_BYTES (QT*S_*4 + QT*S_*2 + QT*16*4)

__global__ __launch_bounds__(256, 1)
void attn_fused_kernel(const float* __restrict__ Q,
                       const float* __restrict__ K,
                       const float* __restrict__ V,
                       const unsigned char* __restrict__ mask,   // [B,1,S] bools
                       const float* __restrict__ rel,            // [1,S,S]
                       float* __restrict__ outP,                 // [B,S,D]
                       float* __restrict__ wP) {                 // [B,S,S]
    extern __shared__ char smem[];
    float*     sc  = (float*)smem;                          // [QT][S_] fp32 scores
    _Float16*  wb  = (_Float16*)(smem + QT*S_*4);           // [QT][S_] fp16 weights
    float*     red = (float*)(smem + QT*S_*4 + QT*S_*2);    // [QT][16]

    const int b    = blockIdx.y;
    const int q0   = blockIdx.x * QT;
    const int tid  = threadIdx.x;
    const int wave = tid >> 5;          // 8 waves (wave32)
    const int lane = tid & 31;
    const int m    = lane & 15;         // M (A) / N (B,C) index
    const int half = lane >> 4;         // lane group 0/1

    const float scale = 0.08838834764831845f;  // 1/sqrt(128)

    // ================= Phase 1: scores = scale*Q.K^T + rel (+mask) -> LDS =================
    // A fragments (Q rows), K-dim chunks of 32 over D=128.
    // A layout: M = lane%16, elem e -> k = (e/8)*16 + (lane/16)*8 + (e%8)  (+32*c)
    v16h aQ[4];
    {
        const float* qrow = Q + (size_t)(b * S_ + q0 + m) * D_;
        #pragma unroll
        for (int c = 0; c < 4; ++c) {
            v16f tf;
            #pragma unroll
            for (int e = 0; e < 16; ++e) {
                int d = c * 32 + (e >> 3) * 16 + half * 8 + (e & 7);
                tf[e] = qrow[d];                 // two contiguous 8-float runs
            }
            aQ[c] = __builtin_convertvector(tf, v16h);  // packed v_cvt_pk_f16_f32
        }
    }

    // Each wave handles key tiles kt = wave, wave+8, ... (16 tiles of 16 keys)
    for (int kt = wave; kt < S_ / 16; kt += 8) {
        const int k0 = kt * 16;
        const float* krow = K + (size_t)(b * S_ + k0 + m) * D_;  // key = k0 + (lane%16)

        // Two independent accumulation chains -> no back-to-back WMMA RAW/WAR stalls.
        v8f acc0 = {}, acc1 = {};
        #pragma unroll
        for (int cc = 0; cc < 2; ++cc) {
            // B layout: N = lane%16, elem e -> k = (e/8)*16 + (lane/16)*8 + (e%8)  (+32*c)
            v16f tf0, tf1;
            #pragma unroll
            for (int e = 0; e < 16; ++e) {
                int d0 = (2*cc + 0) * 32 + (e >> 3) * 16 + half * 8 + (e & 7);   // two contiguous 8-float runs
                int d1 = (2*cc + 1) * 32 + (e >> 3) * 16 + half * 8 + (e & 7);
                tf0[e] = krow[d0];
                tf1[e] = krow[d1];
            }
            v16h bK0 = __builtin_convertvector(tf0, v16h);
            v16h bK1 = __builtin_convertvector(tf1, v16h);
            acc0 = __builtin_amdgcn_wmma_f32_16x16x32_f16(
                false, aQ[2*cc + 0], false, bK0, (short)0, acc0, false, false);
            acc1 = __builtin_amdgcn_wmma_f32_16x16x32_f16(
                false, aQ[2*cc + 1], false, bK1, (short)0, acc1, false, false);
        }

        // C layout: vgpr r -> M = r + 8*(lane/16), N = lane%16
        const bool mvalid = mask[(size_t)b * S_ + k0 + m] != 0;
        #pragma unroll
        for (int r = 0; r < 8; ++r) {
            int M = r + half * 8;
            float s = (acc0[r] + acc1[r]) * scale + rel[(size_t)(q0 + M) * S_ + k0 + m];
            if (!mvalid) s = -INFINITY;
            sc[M * S_ + k0 + m] = s;
        }
    }
    __syncthreads();

    // ================= Phase 2: row softmax (fp32) + stream weights to HBM =================
    {
        const int r  = tid >> 4;   // row 0..15
        const int c0 = tid & 15;   // 16 threads per row
        float mx = -INFINITY;
        for (int j = c0; j < S_; j += 16) mx = fmaxf(mx, sc[r * S_ + j]);
        red[r * 16 + c0] = mx;
        __syncthreads();
        float rowmax = -INFINITY;
        #pragma unroll
        for (int t = 0; t < 16; ++t) rowmax = fmaxf(rowmax, red[r * 16 + t]);
        __syncthreads();

        float sum = 0.f;
        for (int j = c0; j < S_; j += 16) {
            float e = __expf(sc[r * S_ + j] - rowmax);
            sc[r * S_ + j] = e;
            sum += e;
        }
        red[r * 16 + c0] = sum;
        __syncthreads();
        float rowsum = 0.f;
        #pragma unroll
        for (int t = 0; t < 16; ++t) rowsum += red[r * 16 + t];
        const float inv = 1.f / rowsum;

        float* wrow = wP + (size_t)b * S_ * S_ + (size_t)(q0 + r) * S_;
        for (int j = c0; j < S_; j += 16) {
            float w = sc[r * S_ + j] * inv;
            wrow[j] = w;                      // 134 MB HBM store (the bound)
            wb[r * S_ + j] = (_Float16)w;     // fp16 copy for PV GEMM
        }
    }
    __syncthreads();

    // ================= Phase 3: out = W @ V  (wave owns 16 output dims) =================
    {
        const int n0 = wave * 16;       // output d-columns [n0, n0+16)
        // Two independent accumulator chains, unroll-by-2 over K chunks of 32.
        v8f oacc0 = {}, oacc1 = {};
        for (int kc = 0; kc < S_; kc += 64) {
            v16h aW0, aW1;
            #pragma unroll
            for (int e = 0; e < 16; ++e) {
                int koff = (e >> 3) * 16 + half * 8 + (e & 7);
                aW0[e] = wb[m * S_ + kc +  0 + koff];
                aW1[e] = wb[m * S_ + kc + 32 + koff];
            }
            v16f tf0, tf1;
            #pragma unroll
            for (int e = 0; e < 16; ++e) {
                int key = kc + (e >> 3) * 16 + half * 8 + (e & 7);
                tf0[e] = V[(size_t)(b * S_ + key +  0) * D_ + n0 + m];
                tf1[e] = V[(size_t)(b * S_ + key + 32) * D_ + n0 + m];
            }
            v16h bV0 = __builtin_convertvector(tf0, v16h);
            v16h bV1 = __builtin_convertvector(tf1, v16h);
            oacc0 = __builtin_amdgcn_wmma_f32_16x16x32_f16(
                false, aW0, false, bV0, (short)0, oacc0, false, false);
            oacc1 = __builtin_amdgcn_wmma_f32_16x16x32_f16(
                false, aW1, false, bV1, (short)0, oacc1, false, false);
        }
        #pragma unroll
        for (int r = 0; r < 8; ++r) {
            int M = r + half * 8;
            outP[(size_t)(b * S_ + q0 + M) * D_ + n0 + m] = oacc0[r] + oacc1[r];
        }
    }
}

extern "C" void kernel_launch(void* const* d_in, const int* in_sizes, int n_in,
                              void* d_out, int out_size, void* d_ws, size_t ws_size,
                              hipStream_t stream) {
    (void)in_sizes; (void)n_in; (void)out_size; (void)d_ws; (void)ws_size;
    const float*         Q    = (const float*)d_in[0];
    const float*         K    = (const float*)d_in[1];
    const float*         V    = (const float*)d_in[2];
    const unsigned char* mask = (const unsigned char*)d_in[3];
    const float*         rel  = (const float*)d_in[4];

    float* outP = (float*)d_out;                       // [B,S,D] first
    float* wP   = outP + (size_t)B_ * S_ * D_;         // then [B,S,S] weights

    // Allow >64KB dynamic LDS (host-side attribute; deterministic, capture-safe).
    (void)hipFuncSetAttribute((const void*)attn_fused_kernel,
                              hipFuncAttributeMaxDynamicSharedMemorySize, SMEM_BYTES);

    dim3 grid(S_ / QT, B_);   // 128 x 8 workgroups
    attn_fused_kernel<<<grid, 256, SMEM_BYTES, stream>>>(Q, K, V, mask, rel, outP, wP);
}
